// AttentionLayer_68968584839431
// MI455X (gfx1250) — compile-verified, matrix operand order checked by reading, not run on hardware
//
#include <hip/hip_runtime.h>

// Problem dims (fixed by the reference)
#define B_DIM 32
#define N_DIM 2048
#define E_DIM 1024
#define H_DIM 1024

typedef __attribute__((ext_vector_type(16))) __bf16 v16bf;
typedef __attribute__((ext_vector_type(8)))  float  v8f;
typedef __attribute__((ext_vector_type(4)))  unsigned int u32x4;

union Frag16 {
    u32x4 u[2];   // two 16-byte chunks
    v16bf v;      // WMMA operand (16 bf16 per lane)
};

__device__ __forceinline__ unsigned short f2bf(float f) {
    union { float f; unsigned int u; } x; x.f = f;
    // round-to-nearest-even bf16
    unsigned int r = x.u + 0x7FFFu + ((x.u >> 16) & 1u);
    return (unsigned short)(r >> 16);
}

// ---------------------------------------------------------------------------
// Kernel A: w_eT[h][e] = bf16(w_e[e][h])   (2 MB, stays L2-resident)
// ---------------------------------------------------------------------------
__global__ __launch_bounds__(256) void transpose_we_kernel(
    const float* __restrict__ w_e, unsigned short* __restrict__ weT) {
    const int idx = blockIdx.x * 256 + threadIdx.x;   // E*H = 1<<20 threads
    const int h = idx >> 10;          // row of weT
    const int e = idx & (E_DIM - 1);  // col of weT
    weT[idx] = f2bf(w_e[(size_t)e * H_DIM + h]);
}

// ---------------------------------------------------------------------------
// Kernel B: t[b][h] = sum_d dec[b][d] * w_d[d][h]    (67 MFLOP, trivial)
// ---------------------------------------------------------------------------
__global__ __launch_bounds__(256) void dec_proj_kernel(
    const float* __restrict__ dec, const float* __restrict__ w_d,
    float* __restrict__ t) {
    const int idx = blockIdx.x * 256 + threadIdx.x;   // B*H = 32768 threads
    const int b = idx >> 10;
    const int h = idx & (H_DIM - 1);
    float acc = 0.0f;
    for (int d = 0; d < 1024; ++d)
        acc = fmaf(dec[b * 1024 + d], w_d[(size_t)d * H_DIM + h], acc);
    t[idx] = acc;
}

// ---------------------------------------------------------------------------
// Kernel C: fused  scores[row] = w_out . tanh(enc[row,:] @ w_e + t[b,:])
//   One WG (256 thr, 8 waves) per 16 rows. enc tile staged in LDS as bf16.
//   Each wave covers H-tiles {wave, wave+8, ...}; per H-tile: 32x
//   v_wmma_f32_16x16x32_bf16 over E, then tanh/w_out epilogue.
// ---------------------------------------------------------------------------
__global__ __launch_bounds__(256) void scores_kernel(
    const float* __restrict__ enc,
    const unsigned short* __restrict__ weT,   // bf16 bits, [H][E]
    const float* __restrict__ t,              // [B][H]
    const float* __restrict__ w_out,          // [H]
    float* __restrict__ scores)               // [B*N] (probs region of d_out)
{
    __shared__ __align__(16) unsigned short Atile[16 * E_DIM]; // 32 KB bf16
    __shared__ float s_part[8][16];                            // per-wave rows

    const int rowbase = blockIdx.x * 16;        // 4096 blocks -> 65536 rows
    const int b       = rowbase >> 11;          // rows per batch = N_DIM=2048
    const int tid     = threadIdx.x;

    // Stage 16 enc rows (fp32 -> bf16) into LDS, coalesced on e.
    for (int i = tid; i < 16 * E_DIM; i += 256) {
        const int r = i >> 10;
        const int e = i & (E_DIM - 1);
        Atile[i] = f2bf(enc[(size_t)(rowbase + r) * E_DIM + e]);
    }
    __syncthreads();

    const int wave = tid >> 5;     // 0..7
    const int lane = tid & 31;
    const int l15  = lane & 15;    // A: row index; B/C: column index
    const int kh   = lane >> 4;    // which K-half of the 16-bit fragment

    float acc[8];
#pragma unroll
    for (int r = 0; r < 8; ++r) acc[r] = 0.0f;

    for (int ht = wave; ht < H_DIM / 16; ht += 8) {
        const int h0 = ht * 16;
        v8f c = {0.f, 0.f, 0.f, 0.f, 0.f, 0.f, 0.f, 0.f};

        for (int k0 = 0; k0 < E_DIM; k0 += 32) {
            // A fragment from LDS: lane = row l15, K-chunks at k0+8*kh and +16
            Frag16 a, bm;
            const unsigned short* arow = &Atile[l15 * E_DIM];
            a.u[0] = *(const u32x4*)&arow[k0 + 8 * kh];
            a.u[1] = *(const u32x4*)&arow[k0 + 16 + 8 * kh];
            // B fragment from global (L2-hot w_eT): column n = l15 -> row h0+l15
            const unsigned short* brow = &weT[(size_t)(h0 + l15) * E_DIM];
            bm.u[0] = *(const u32x4*)&brow[k0 + 8 * kh];
            bm.u[1] = *(const u32x4*)&brow[k0 + 16 + 8 * kh];

            c = __builtin_amdgcn_wmma_f32_16x16x32_bf16(
                    false, a.v, false, bm.v, (short)0, c, false, false);
        }

        // Epilogue: bias + tanh + dot with w_out (column = l15 within tile).
        const float bias = t[b * H_DIM + h0 + l15];
        const float wo   = w_out[h0 + l15];
#pragma unroll
        for (int r = 0; r < 8; ++r)
            acc[r] += tanhf(c[r] + bias) * wo;   // row = r + 8*kh, col = l15
    }

    // Reduce over the 16 columns held by lanes sharing the same K-half.
#pragma unroll
    for (int r = 0; r < 8; ++r) {
        acc[r] += __shfl_xor(acc[r], 1, 32);
        acc[r] += __shfl_xor(acc[r], 2, 32);
        acc[r] += __shfl_xor(acc[r], 4, 32);
        acc[r] += __shfl_xor(acc[r], 8, 32);
    }
    if (l15 == 0) {       // lanes 0 (rows 0..7) and 16 (rows 8..15)
#pragma unroll
        for (int r = 0; r < 8; ++r)
            s_part[wave][r + 8 * kh] = acc[r];
    }
    __syncthreads();

    // Deterministic cross-wave sum (fixed order, no float atomics).
    if (tid < 16) {
        float s = 0.0f;
#pragma unroll
        for (int w = 0; w < 8; ++w) s += s_part[w][tid];
        scores[rowbase + tid] = s;
    }
}

// ---------------------------------------------------------------------------
// Kernel D: in-place masked softmax over N per batch row.
// ---------------------------------------------------------------------------
__global__ __launch_bounds__(256) void softmax_kernel(
    float* __restrict__ probs, const unsigned char* __restrict__ mask) {
    const int b   = blockIdx.x;
    const int tid = threadIdx.x;
    __shared__ float red[256];

    float v[8];
    float m = -__builtin_inff();
#pragma unroll
    for (int i = 0; i < 8; ++i) {
        const int n = tid + i * 256;
        float s = probs[b * N_DIM + n];
        s = mask[b * N_DIM + n] ? s : -__builtin_inff();
        v[i] = s;
        m = fmaxf(m, s);
    }
    red[tid] = m; __syncthreads();
    for (int off = 128; off > 0; off >>= 1) {
        if (tid < off) red[tid] = fmaxf(red[tid], red[tid + off]);
        __syncthreads();
    }
    m = red[0]; __syncthreads();

    float sum = 0.0f;
#pragma unroll
    for (int i = 0; i < 8; ++i) { v[i] = expf(v[i] - m); sum += v[i]; }
    red[tid] = sum; __syncthreads();
    for (int off = 128; off > 0; off >>= 1) {
        if (tid < off) red[tid] += red[tid + off];
        __syncthreads();
    }
    const float inv = 1.0f / red[0];
#pragma unroll
    for (int i = 0; i < 8; ++i)
        probs[b * N_DIM + tid + i * 256] = v[i] * inv;
}

// ---------------------------------------------------------------------------
// Kernel E: attn[b][e] = sum_n probs[b][n] * enc[b][n][e]   (HBM-bound pass 2)
// ---------------------------------------------------------------------------
__global__ __launch_bounds__(256) void attn_kernel(
    const float* __restrict__ enc, const float* __restrict__ probs,
    float* __restrict__ attn) {
    __shared__ float p[256];
    const int b = blockIdx.x >> 2;                   // E/256 = 4 chunks per b
    const int e = (blockIdx.x & 3) * 256 + threadIdx.x;

    float acc = 0.0f;
    for (int n0 = 0; n0 < N_DIM; n0 += 256) {
        __syncthreads();
        p[threadIdx.x] = probs[b * N_DIM + n0 + threadIdx.x];
        __syncthreads();
        if (n0 + 256 < N_DIM)   // hint: pull next tile toward L2 early
            __builtin_prefetch(&enc[((size_t)b * N_DIM + n0 + 256) * E_DIM + e], 0, 0);
        for (int j = 0; j < 256; ++j)
            acc = fmaf(p[j], enc[((size_t)b * N_DIM + n0 + j) * E_DIM + e], acc);
    }
    attn[b * E_DIM + e] = acc;
}

// ---------------------------------------------------------------------------
extern "C" void kernel_launch(void* const* d_in, const int* in_sizes, int n_in,
                              void* d_out, int out_size, void* d_ws, size_t ws_size,
                              hipStream_t stream) {
    (void)in_sizes; (void)n_in; (void)out_size; (void)ws_size;

    const float*         enc   = (const float*)d_in[0];        // [B,N,E]
    const float*         dec   = (const float*)d_in[1];        // [B,D]
    const unsigned char* imask = (const unsigned char*)d_in[2];// [B,N] bool
    const float*         w_e   = (const float*)d_in[3];        // [E,H]
    const float*         w_d   = (const float*)d_in[4];        // [D,H]
    const float*         w_out = (const float*)d_in[5];        // [H,1]

    float* out   = (float*)d_out;
    float* attn  = out;                     // [B,E] = 32768 floats
    float* probs = out + B_DIM * E_DIM;     // [B,N] = 65536 floats

    // Workspace: bf16 w_eT (2 MB) + t (128 KB)
    unsigned short* weT = (unsigned short*)d_ws;
    float* t = (float*)((char*)d_ws + (size_t)E_DIM * H_DIM * sizeof(unsigned short));

    transpose_we_kernel<<<(E_DIM * H_DIM) / 256, 256, 0, stream>>>(w_e, weT);
    dec_proj_kernel<<<(B_DIM * H_DIM) / 256, 256, 0, stream>>>(dec, w_d, t);
    scores_kernel<<<(B_DIM * N_DIM) / 16, 256, 0, stream>>>(enc, weT, t, w_out, probs);
    softmax_kernel<<<B_DIM, 256, 0, stream>>>(probs, imask);
    attn_kernel<<<B_DIM * (E_DIM / 256), 256, 0, stream>>>(enc, probs, attn);
}
